// InterpretableMultiHeadAttention_8899172237368
// MI455X (gfx1250) — matrix operand order repaired and checked by reading, not compiled, not run on hardware
//
#include <hip/hip_runtime.h>
#include <stdint.h>

// ---------------------------------------------------------------------------
// Problem constants (match the reference)
// ---------------------------------------------------------------------------
#define TSEQ   2048
#define DMODEL 512
#define NHEADS 8
#define DKHEAD 64
#define NBATCH 4

// ---------------------------------------------------------------------------
// WMMA tiling: block = 256 threads = 8 wave32, wave grid 4(m) x 2(n)
// macro tile 128(M) x 64(N) x 32(K); each wave computes 32x32 = 2x2 frags.
// Double-buffered LDS tiles: 2 * (12KB + 6KB) = 36KB per workgroup.
// ---------------------------------------------------------------------------
#define BM 128
#define BN 64
#define BK 32
#define LDSA 48   // A-tile row stride (elements): 96B rows, 16B aligned, skewed
#define LDSB 48
#define NTHREADS 256

typedef __attribute__((ext_vector_type(16))) __bf16 v16bf;
typedef __attribute__((ext_vector_type(8)))  float  v8f;

struct Frag { union { v16bf v; uint4 u[2]; }; };
union Half8 { uint4 u; __bf16 e[8]; };

// ---------------------------------------------------------------------------
// CDNA5 async global->LDS copies (ASYNCcnt-tracked, bypass VGPRs).
// Probe result: builtin exists, signature is
//   void __builtin_amdgcn_global_load_async_to_lds_b128(v4i AS1*, v4i AS3*,
//                                                       imm offset, imm cpol)
// Guarded: fall back to b128 register copies if the builtin is absent.
// ---------------------------------------------------------------------------
#if defined(__HIP_DEVICE_COMPILE__) && \
    __has_builtin(__builtin_amdgcn_global_load_async_to_lds_b128)
#define HAVE_ASYNC_LDS 1
typedef __attribute__((ext_vector_type(4))) int v4i;
typedef __attribute__((address_space(1))) v4i as1_v4i;
typedef __attribute__((address_space(3))) v4i as3_v4i;
#else
#define HAVE_ASYNC_LDS 0
#endif

// copy 16B (8 bf16) global -> LDS
__device__ __forceinline__ void copy16_g2l(const __bf16* g, __bf16* l) {
#if HAVE_ASYNC_LDS
  __builtin_amdgcn_global_load_async_to_lds_b128((as1_v4i*)g, (as3_v4i*)l, 0,
                                                 0);
#else
  *(uint4*)l = *(const uint4*)g;
#endif
}

__device__ __forceinline__ void async_wait0() {
#if HAVE_ASYNC_LDS
#if __has_builtin(__builtin_amdgcn_s_wait_asynccnt)
  __builtin_amdgcn_s_wait_asynccnt(0);
#else
  asm volatile("s_wait_asynccnt 0x0" ::: "memory");
#endif
#endif
}

// ---------------------------------------------------------------------------
// Core WMMA step: one 128x64x32 macro-tile MAC from LDS tiles.
// A fragment (16x32 bf16): lane<16 row=lane K=[0..7],[16..23];
//                          lane>=16 row=lane-16 K=[8..15],[24..31]
// B fragment (32x16 bf16): lane&15 = N col; lane<16 K=[0..7],[16..23], lane>=16 K=[8..15],[24..31]
// ---------------------------------------------------------------------------
__device__ __forceinline__ void mma_step(const __bf16* As, const __bf16* Bs,
                                         v8f acc[2][2]) {
  const int lane = threadIdx.x & 31;
  const int wave = threadIdx.x >> 5;
  const int wm = wave >> 1, wn = wave & 1;
  const int half = lane >> 4, lr = lane & 15;
  Frag a[2], b[2];
#pragma unroll
  for (int mi = 0; mi < 2; ++mi) {
    const __bf16* ap = As + (wm * 32 + mi * 16 + lr) * LDSA;
    a[mi].u[0] = *(const uint4*)(ap + half * 8);
    a[mi].u[1] = *(const uint4*)(ap + 16 + half * 8);
  }
#pragma unroll
  for (int ni = 0; ni < 2; ++ni) {
    const __bf16* bp = Bs + (wn * 32 + ni * 16 + lr) * LDSB;
    b[ni].u[0] = *(const uint4*)(bp + half * 8);
    b[ni].u[1] = *(const uint4*)(bp + 16 + half * 8);
  }
#pragma unroll
  for (int mi = 0; mi < 2; ++mi)
#pragma unroll
    for (int ni = 0; ni < 2; ++ni)
      acc[mi][ni] = __builtin_amdgcn_wmma_f32_16x16x32_bf16(
          false, a[mi].v, false, b[ni].v, (short)0, acc[mi][ni], false, false);
}

__device__ __forceinline__ void zero_acc(v8f acc[2][2]) {
#pragma unroll
  for (int mi = 0; mi < 2; ++mi)
#pragma unroll
    for (int ni = 0; ni < 2; ++ni)
#pragma unroll
      for (int e = 0; e < 8; ++e) acc[mi][ni][e] = 0.0f;
}

// ---------------------------------------------------------------------------
// Global -> LDS tile loaders (256 threads). All vector ops are 16B aligned.
// ---------------------------------------------------------------------------
// A tile 128x32, f32 source -> bf16 LDS (convert en route; VGPR path).
__device__ __forceinline__ void load_a_f32(const float* A, int lda, __bf16* As) {
  const int t = threadIdx.x;
  const int row = t >> 1;
  const int kq = (t & 1) * 16;
  const float4* s = (const float4*)(A + (size_t)row * lda + kq);
  alignas(16) float f[16];
  *(float4*)&f[0]  = s[0];
  *(float4*)&f[4]  = s[1];
  *(float4*)&f[8]  = s[2];
  *(float4*)&f[12] = s[3];
  Frag tmp;
  __bf16* e = (__bf16*)&tmp;
#pragma unroll
  for (int i = 0; i < 16; ++i) e[i] = (__bf16)f[i];
  uint4* d = (uint4*)(As + row * LDSA + kq);
  d[0] = tmp.u[0];
  d[1] = tmp.u[1];
}

// A tile 128x32, bf16 source -> bf16 LDS (async copy path).
__device__ __forceinline__ void load_a_bf16(const __bf16* A, int lda, __bf16* As) {
  const int t = threadIdx.x;
  const int row = t >> 1;
  const int kq = (t & 1) * 16;
  const __bf16* g = A + (size_t)row * lda + kq;
  __bf16* l = As + row * LDSA + kq;
  copy16_g2l(g, l);
  copy16_g2l(g + 8, l + 8);
}

// B tile 64x32, f32 source -> bf16 LDS (convert en route; VGPR path).
__device__ __forceinline__ void load_b_f32(const float* B, int ldb, __bf16* Bs) {
  const int t = threadIdx.x;
  const int row = t >> 2;
  const int kq = (t & 3) * 8;
  const float4* s = (const float4*)(B + (size_t)row * ldb + kq);
  alignas(16) float f[8];
  *(float4*)&f[0] = s[0];
  *(float4*)&f[4] = s[1];
  Half8 h8;
#pragma unroll
  for (int i = 0; i < 8; ++i) h8.e[i] = (__bf16)f[i];
  *(uint4*)(Bs + row * LDSB + kq) = h8.u;
}

// B tile 64x32, bf16 source -> bf16 LDS (async copy path).
__device__ __forceinline__ void load_b_bf16(const __bf16* B, int ldb, __bf16* Bs) {
  const int t = threadIdx.x;
  const int row = t >> 2;
  const int kq = (t & 3) * 8;
  copy16_g2l(B + (size_t)row * ldb + kq, Bs + row * LDSB + kq);
}

// A tile 128x32 built as head-average of attention: (1/8) * sum_h attn[h].
// Ab points at attn[b, 0, m0+., kk]; head stride = TSEQ*TSEQ f32.
__device__ __forceinline__ void load_a_headsum(const float* Ab, __bf16* As) {
  const int t = threadIdx.x;
  const int row = t >> 1;
  const int kq = (t & 1) * 16;
  const float* base = Ab + (size_t)row * TSEQ + kq;
  float sum[16];
#pragma unroll
  for (int i = 0; i < 16; ++i) sum[i] = 0.0f;
#pragma unroll
  for (int h = 0; h < NHEADS; ++h) {
    const float4* p = (const float4*)(base + (size_t)h * TSEQ * TSEQ);
    alignas(16) float f[16];
    *(float4*)&f[0]  = p[0];
    *(float4*)&f[4]  = p[1];
    *(float4*)&f[8]  = p[2];
    *(float4*)&f[12] = p[3];
#pragma unroll
    for (int i = 0; i < 16; ++i) sum[i] += f[i];
  }
  Frag tmp;
  __bf16* e = (__bf16*)&tmp;
#pragma unroll
  for (int i = 0; i < 16; ++i) e[i] = (__bf16)(sum[i] * 0.125f);
  uint4* d = (uint4*)(As + row * LDSA + kq);
  d[0] = tmp.u[0];
  d[1] = tmp.u[1];
}

// ---------------------------------------------------------------------------
// Kernel 1: projection GEMM  C[M,N] = X[M,K] @ W[N,K]^T + bias, bf16 output.
// Double-buffered LDS pipeline.
// ---------------------------------------------------------------------------
__global__ __launch_bounds__(NTHREADS) void k_proj_bf16(
    const float* __restrict__ X, const float* __restrict__ W,
    const float* __restrict__ bias, __bf16* __restrict__ C, int N, int K) {
  __shared__ __bf16 As[2][BM * LDSA];
  __shared__ __bf16 Bs[2][BN * LDSB];
  const int m0 = blockIdx.x * BM;
  const int n0 = blockIdx.y * BN;
  v8f acc[2][2];
  zero_acc(acc);
  const int nsteps = K / BK;
  load_a_f32(X + (size_t)m0 * K, K, As[0]);
  load_b_f32(W + (size_t)n0 * K, K, Bs[0]);
  __syncthreads();
  for (int s = 0; s < nsteps; ++s) {
    const int cur = s & 1, nxt = cur ^ 1;
    if (s + 1 < nsteps) {
      load_a_f32(X + (size_t)m0 * K + (s + 1) * BK, K, As[nxt]);
      load_b_f32(W + (size_t)n0 * K + (s + 1) * BK, K, Bs[nxt]);
    }
    mma_step(As[cur], Bs[cur], acc);
    __syncthreads();
  }
  const int lane = threadIdx.x & 31;
  const int wave = threadIdx.x >> 5;
  const int wm = wave >> 1, wn = wave & 1;
  const int half = lane >> 4, lr = lane & 15;
#pragma unroll
  for (int mi = 0; mi < 2; ++mi)
#pragma unroll
    for (int ni = 0; ni < 2; ++ni) {
      const int col = n0 + wn * 32 + ni * 16 + lr;
      const float bv = bias[col];
#pragma unroll
      for (int r = 0; r < 8; ++r) {
        const int row = m0 + wm * 32 + mi * 16 + half * 8 + r;
        C[(size_t)row * N + col] = (__bf16)(acc[mi][ni][r] + bv);
      }
    }
}

// ---------------------------------------------------------------------------
// Kernel 2: bf16 tiled transpose  Vt[b, d, t] = Vb[b*T + t, d]
// ---------------------------------------------------------------------------
__global__ __launch_bounds__(NTHREADS) void k_transpose(
    const __bf16* __restrict__ Vb, __bf16* __restrict__ Vt) {
  __shared__ __bf16 tile[32][33];
  const int b = blockIdx.z;
  const int d0 = blockIdx.x * 32;
  const int t0 = blockIdx.y * 32;
  const int lx = threadIdx.x & 31;
  const int ly = threadIdx.x >> 5;  // 0..7
#pragma unroll
  for (int i = 0; i < 32; i += 8)
    tile[ly + i][lx] =
        Vb[((size_t)b * TSEQ + t0 + ly + i) * DMODEL + d0 + lx];
  __syncthreads();
#pragma unroll
  for (int i = 0; i < 32; i += 8)
    Vt[((size_t)b * DMODEL + d0 + ly + i) * TSEQ + t0 + lx] = tile[lx][ly + i];
}

// ---------------------------------------------------------------------------
// Kernel 3: raw scores  S[b,h] = (Q[b,h] @ K[b,h]^T) / sqrt(dk), f32 into the
// attn region of d_out (softmaxed in place by kernel 4). Async LDS staging.
// ---------------------------------------------------------------------------
__global__ __launch_bounds__(NTHREADS) void k_scores(
    const __bf16* __restrict__ Qb, const __bf16* __restrict__ Kb,
    float* __restrict__ attn) {
  __shared__ __bf16 As[2][BM * LDSA];
  __shared__ __bf16 Bs[2][BN * LDSB];
  const int bh = blockIdx.z;
  const int b = bh >> 3, h = bh & 7;
  const int m0 = blockIdx.x * BM;
  const int n0 = blockIdx.y * BN;
  const __bf16* Aq = Qb + ((size_t)b * TSEQ + m0) * DMODEL + h * DKHEAD;
  const __bf16* Bk = Kb + ((size_t)b * TSEQ + n0) * DMODEL + h * DKHEAD;
  float* Cp = attn + (size_t)bh * TSEQ * TSEQ;
  v8f acc[2][2];
  zero_acc(acc);
  const int nsteps = DKHEAD / BK;  // 2
  load_a_bf16(Aq, DMODEL, As[0]);
  load_b_bf16(Bk, DMODEL, Bs[0]);
  async_wait0();
  __syncthreads();
#pragma unroll
  for (int s = 0; s < nsteps; ++s) {
    const int cur = s & 1, nxt = cur ^ 1;
    if (s + 1 < nsteps) {
      load_a_bf16(Aq + (s + 1) * BK, DMODEL, As[nxt]);
      load_b_bf16(Bk + (s + 1) * BK, DMODEL, Bs[nxt]);
    }
    mma_step(As[cur], Bs[cur], acc);
    async_wait0();
    __syncthreads();
  }
  const int lane = threadIdx.x & 31;
  const int wave = threadIdx.x >> 5;
  const int wm = wave >> 1, wn = wave & 1;
  const int half = lane >> 4, lr = lane & 15;
#pragma unroll
  for (int mi = 0; mi < 2; ++mi)
#pragma unroll
    for (int ni = 0; ni < 2; ++ni) {
      const int cl = n0 + wn * 32 + ni * 16 + lr;
#pragma unroll
      for (int r = 0; r < 8; ++r) {
        const int rl = m0 + wm * 32 + mi * 16 + half * 8 + r;
        Cp[(size_t)rl * TSEQ + cl] = acc[mi][ni][r] * 0.125f;  // 1/sqrt(64)
      }
    }
}

// ---------------------------------------------------------------------------
// Kernel 4: in-place row softmax over 2048-wide rows; one wave32 per row.
// ---------------------------------------------------------------------------
__global__ __launch_bounds__(NTHREADS) void k_softmax(float* __restrict__ attn) {
  const int wave = threadIdx.x >> 5;
  const int lane = threadIdx.x & 31;
  const size_t row = (size_t)blockIdx.x * 8 + wave;
  float* p = attn + row * TSEQ;
  float v[64];
  float m = -3.4e38f;
#pragma unroll
  for (int i = 0; i < 64; ++i) {
    v[i] = p[lane + i * 32];
    m = fmaxf(m, v[i]);
  }
#pragma unroll
  for (int o = 16; o > 0; o >>= 1) m = fmaxf(m, __shfl_xor(m, o, 32));
  float s = 0.0f;
#pragma unroll
  for (int i = 0; i < 64; ++i) {
    v[i] = __expf(v[i] - m);
    s += v[i];
  }
#pragma unroll
  for (int o = 16; o > 0; o >>= 1) s += __shfl_xor(s, o, 32);
  const float inv = 1.0f / s;
#pragma unroll
  for (int i = 0; i < 64; ++i) p[lane + i * 32] = v[i] * inv;
}

// ---------------------------------------------------------------------------
// Kernel 5: ctx[b] = ((1/H) sum_h attn[b,h]) @ V[b], bf16 output.
// B tiles async-copied from transposed V; A tiles head-summed on the fly.
// ---------------------------------------------------------------------------
__global__ __launch_bounds__(NTHREADS) void k_ctx(
    const float* __restrict__ attn, const __bf16* __restrict__ Vt,
    __bf16* __restrict__ Ctx) {
  __shared__ __bf16 As[2][BM * LDSA];
  __shared__ __bf16 Bs[2][BN * LDSB];
  const int b = blockIdx.z;
  const int m0 = blockIdx.x * BM;  // query rows
  const int n0 = blockIdx.y * BN;  // model dim
  const float* Ab =
      attn + (size_t)b * NHEADS * TSEQ * TSEQ + (size_t)m0 * TSEQ;
  const __bf16* Bv = Vt + ((size_t)b * DMODEL + n0) * TSEQ;
  v8f acc[2][2];
  zero_acc(acc);
  const int nsteps = TSEQ / BK;  // 64
  load_b_bf16(Bv, TSEQ, Bs[0]);  // async B first, overlap with A math
  load_a_headsum(Ab, As[0]);
  async_wait0();
  __syncthreads();
  for (int s = 0; s < nsteps; ++s) {
    const int cur = s & 1, nxt = cur ^ 1;
    if (s + 1 < nsteps) {
      load_b_bf16(Bv + (s + 1) * BK, TSEQ, Bs[nxt]);
      load_a_headsum(Ab + (s + 1) * BK, As[nxt]);
    }
    mma_step(As[cur], Bs[cur], acc);
    async_wait0();
    __syncthreads();
  }
  const int lane = threadIdx.x & 31;
  const int wave = threadIdx.x >> 5;
  const int wm = wave >> 1, wn = wave & 1;
  const int half = lane >> 4, lr = lane & 15;
#pragma unroll
  for (int mi = 0; mi < 2; ++mi)
#pragma unroll
    for (int ni = 0; ni < 2; ++ni) {
      const int cl = n0 + wn * 32 + ni * 16 + lr;
#pragma unroll
      for (int r = 0; r < 8; ++r) {
        const int rl = m0 + wm * 32 + mi * 16 + half * 8 + r;
        Ctx[((size_t)b * TSEQ + rl) * DMODEL + cl] = (__bf16)acc[mi][ni][r];
      }
    }
}

// ---------------------------------------------------------------------------
// Kernel 6: output projection  Out = Ctx @ Wo^T + bo, f32 output.
// A tiles async-copied (bf16); B tiles converted f32->bf16 en route.
// ---------------------------------------------------------------------------
__global__ __launch_bounds__(NTHREADS) void k_out(
    const __bf16* __restrict__ Ctx, const float* __restrict__ Wo,
    const float* __restrict__ bo, float* __restrict__ Out) {
  __shared__ __bf16 As[2][BM * LDSA];
  __shared__ __bf16 Bs[2][BN * LDSB];
  const int m0 = blockIdx.x * BM;
  const int n0 = blockIdx.y * BN;
  v8f acc[2][2];
  zero_acc(acc);
  const int nsteps = DMODEL / BK;  // 16
  load_a_bf16(Ctx + (size_t)m0 * DMODEL, DMODEL, As[0]);
  load_b_f32(Wo + (size_t)n0 * DMODEL, DMODEL, Bs[0]);
  async_wait0();
  __syncthreads();
  for (int s = 0; s < nsteps; ++s) {
    const int cur = s & 1, nxt = cur ^ 1;
    if (s + 1 < nsteps) {
      load_a_bf16(Ctx + (size_t)m0 * DMODEL + (s + 1) * BK, DMODEL, As[nxt]);
      load_b_f32(Wo + (size_t)n0 * DMODEL + (s + 1) * BK, DMODEL, Bs[nxt]);
    }
    mma_step(As[cur], Bs[cur], acc);
    async_wait0();
    __syncthreads();
  }
  const int lane = threadIdx.x & 31;
  const int wave = threadIdx.x >> 5;
  const int wm = wave >> 1, wn = wave & 1;
  const int half = lane >> 4, lr = lane & 15;
#pragma unroll
  for (int mi = 0; mi < 2; ++mi)
#pragma unroll
    for (int ni = 0; ni < 2; ++ni) {
      const int col = n0 + wn * 32 + ni * 16 + lr;
      const float bv = bo[col];
#pragma unroll
      for (int r = 0; r < 8; ++r) {
        const int row = m0 + wm * 32 + mi * 16 + half * 8 + r;
        Out[(size_t)row * DMODEL + col] = acc[mi][ni][r] + bv;
      }
    }
}

// ---------------------------------------------------------------------------
// Host launcher
// ---------------------------------------------------------------------------
extern "C" void kernel_launch(void* const* d_in, const int* in_sizes, int n_in,
                              void* d_out, int out_size, void* d_ws,
                              size_t ws_size, hipStream_t stream) {
  (void)in_sizes; (void)n_in; (void)out_size; (void)ws_size;
  const float* query  = (const float*)d_in[0];
  const float* key_in = (const float*)d_in[1];
  const float* value  = (const float*)d_in[2];
  const float* Wq = (const float*)d_in[3];
  const float* bq = (const float*)d_in[4];
  const float* Wk = (const float*)d_in[5];
  const float* bk = (const float*)d_in[6];
  const float* Wv = (const float*)d_in[7];
  const float* bv = (const float*)d_in[8];
  const float* Wo = (const float*)d_in[9];
  const float* bo = (const float*)d_in[10];

  float* out  = (float*)d_out;                               // [B,T,D]
  float* attn = out + (size_t)NBATCH * TSEQ * DMODEL;        // [B,H,T,T]

  const size_t MTOK = (size_t)NBATCH * TSEQ;                 // 8192
  __bf16* Qb  = (__bf16*)d_ws;                               // [8192,512] bf16
  __bf16* Kb  = Qb  + MTOK * DMODEL;
  __bf16* Vb  = Kb  + MTOK * DMODEL;
  __bf16* Vt  = Vb  + MTOK * DMODEL;                         // [B,512,2048]
  __bf16* Ctx = Vt  + MTOK * DMODEL;                         // [8192,512]

  dim3 blk(NTHREADS);

  // 1) Q/K/V projections (M=8192, N=512, K=512)
  dim3 gproj(MTOK / BM, DMODEL / BN);
  k_proj_bf16<<<gproj, blk, 0, stream>>>(query,  Wq, bq, Qb, DMODEL, DMODEL);
  k_proj_bf16<<<gproj, blk, 0, stream>>>(key_in, Wk, bk, Kb, DMODEL, DMODEL);
  k_proj_bf16<<<gproj, blk, 0, stream>>>(value,  Wv, bv, Vb, DMODEL, DMODEL);

  // 2) V transpose for K-contiguous B-fragments in the ctx GEMM
  dim3 gtr(DMODEL / 32, TSEQ / 32, NBATCH);
  k_transpose<<<gtr, blk, 0, stream>>>(Vb, Vt);

  // 3) raw attention scores into d_out's attn region
  dim3 gsc(TSEQ / BM, TSEQ / BN, NBATCH * NHEADS);
  k_scores<<<gsc, blk, 0, stream>>>(Qb, Kb, attn);

  // 4) in-place softmax (one wave per row, 8 rows per block)
  dim3 gsm((NBATCH * NHEADS * TSEQ) / 8);
  k_softmax<<<gsm, blk, 0, stream>>>(attn);

  // 5) ctx = head-averaged attn @ V
  dim3 gctx(TSEQ / BM, DMODEL / BN, NBATCH);
  k_ctx<<<gctx, blk, 0, stream>>>(attn, Vt, Ctx);

  // 6) output projection
  dim3 gout(MTOK / BM, DMODEL / BN);
  k_out<<<gout, blk, 0, stream>>>(Ctx, Wo, bo, out);
}
